// LGR_80650895884888
// MI455X (gfx1250) — matrix operand order repaired and checked by reading, not compiled, not run on hardware
//
#include <hip/hip_runtime.h>
#include <math.h>

// ---------------- problem constants ----------------
#define B_BATCH 64
#define NPTS    8192
#define DEPTH   12
#define CH      128
#define TILE_N  128          // points per block
#define ROWP    136          // padded LDS row stride (halfs): 272B, 16B aligned, breaks 256B bank period
#define THREADS 256          // 8 wave32s

typedef _Float16 v8h  __attribute__((ext_vector_type(8)));
typedef _Float16 v16h __attribute__((ext_vector_type(16)));
typedef float    v8f  __attribute__((ext_vector_type(8)));

// ---------------- workspace layout ----------------
#define WPACK_ELEMS (2 * DEPTH * CH * CH)   // 393216 halfs = 786432 bytes
#define XWX_OFF     ((size_t)WPACK_ELEMS * 2)
#define XWX_ELEMS   (B_BATCH * 81)

// Pack fp32 weights into f16 fragments laid out exactly as the CDNA5 WMMA
// 16-bit A-matrix expects (16x32 tile; lanes 0-15: K=0..7,16..23; lanes 16-31:
// K=8..15,24..31; 2 halfs per dword). One fragment = 32 lanes * 16 halfs.
__global__ void LGR_pack_weights(const float* __restrict__ W1s,
                                 const float* __restrict__ W2s,
                                 _Float16* __restrict__ wpack) {
    int e = blockIdx.x * 256 + threadIdx.x;
    if (e >= WPACK_ELEMS) return;
    int mat = e >> 14;                 // 0..23  (layer*2 + half)
    int r   = e & 16383;
    int m = r >> 7, k = r & 127;       // row-major [out_ch][in_ch]
    int d = mat >> 1, which = mat & 1;
    float val = which ? W2s[d * CH * CH + m * CH + k]
                      : W1s[d * CH * CH + m * CH + k];
    int mt = m >> 4, mr = m & 15, kt = k >> 5, kk = k & 31;
    int g = kk >> 3;                             // group of 8 K values
    int lane = (g & 1) ? (16 + mr) : mr;
    int j = ((g >> 1) ? 4 : 0) + ((kk & 7) >> 1);
    int pos = kk & 1;
    int dst = mat * 16384 + ((mt << 2) + kt) * 512 + lane * 16 + j * 2 + pos;
    wpack[dst] = (_Float16)val;
}

// Fused MLP + logits + weighted 9x9 Gram accumulation.
__global__ __launch_bounds__(THREADS)
void LGR_mlp_xwx(const float* __restrict__ x,      // (B,4,1,N)
                 const float* __restrict__ Wf,     // (128,4)
                 const float* __restrict__ bfirst, // (128)
                 const float* __restrict__ b1s,    // (12,128)
                 const float* __restrict__ b2s,    // (12,128)
                 const float* __restrict__ Wfin,   // (1,128)
                 const float* __restrict__ bfin,   // (1)
                 const _Float16* __restrict__ wpack,
                 float* __restrict__ xwx)          // (64,81)
{
    __shared__ _Float16 bufA[TILE_N * ROWP];   // h   (point-major, f16)
    __shared__ _Float16 bufB[TILE_N * ROWP];   // y1

    const int tid  = threadIdx.x;
    const int lane = tid & 31;
    const int wave = tid >> 5;
    const int blk  = blockIdx.x;
    const int b    = blk >> 6;                 // 64 tiles per batch
    const int tile = blk & 63;
    const int p0   = tile * TILE_N;
    const float* xb = x + (size_t)b * 4 * NPTS;

    // ---------- first layer: h = W_first @ x + b (K=4, plain VALU) ----------
    {
        int p = tid & 127;
        int chalf = tid >> 7;                  // 0 or 1
        float x0 = xb[0 * NPTS + p0 + p];
        float x1 = xb[1 * NPTS + p0 + p];
        float x2 = xb[2 * NPTS + p0 + p];
        float x3 = xb[3 * NPTS + p0 + p];
        for (int c = chalf * 64; c < chalf * 64 + 64; ++c) {
            float h = bfirst[c] + Wf[c * 4 + 0] * x0 + Wf[c * 4 + 1] * x1
                                + Wf[c * 4 + 2] * x2 + Wf[c * 4 + 3] * x3;
            bufA[p * ROWP + c] = (_Float16)h;
        }
    }
    __syncthreads();

    // ---------- residual blocks via WMMA ----------
    const int mt    = wave;                          // wave owns one 16-row M tile
    const int mBase = mt * 16 + ((lane >= 16) ? 8 : 0);
    const int nSub  = lane & 15;
    const int kHalf = (lane < 16) ? 0 : 8;           // B operand: lanes 0-15 K 0..7,16..23

    for (int d = 0; d < DEPTH; ++d) {
        // ---- half 1: bufB = relu(W1 @ bufA + b1) ----
        {
            const _Float16* wmat = wpack + (size_t)(2 * d + 0) * 16384;
            // prefetch next half-layer's fragments for this wave into L2/L0
            const _Float16* wnext = wpack + (size_t)(2 * d + 1) * 16384;
            #pragma unroll
            for (int kt = 0; kt < 4; ++kt)
                __builtin_prefetch(wnext + ((mt * 4 + kt) * 512) + lane * 16, 0, 0);

            v16h aF[4];
            #pragma unroll
            for (int kt = 0; kt < 4; ++kt)
                aF[kt] = *(const v16h*)(wmat + ((mt * 4 + kt) * 512) + lane * 16);
            float bias[8];
            #pragma unroll
            for (int r = 0; r < 8; ++r) bias[r] = b1s[d * CH + mBase + r];

            for (int nt = 0; nt < 8; ++nt) {
                v8f acc = {};
                #pragma unroll
                for (int kt = 0; kt < 4; ++kt) {
                    const _Float16* src = &bufA[(nt * 16 + nSub) * ROWP + kt * 32 + kHalf];
                    v8h lo = *(const v8h*)(src);
                    v8h hi = *(const v8h*)(src + 16);
                    v16h bF;
                    #pragma unroll
                    for (int i = 0; i < 8; ++i) { bF[i] = lo[i]; bF[8 + i] = hi[i]; }
                    acc = __builtin_amdgcn_wmma_f32_16x16x32_f16(
                        false, aF[kt], false, bF, (short)0, acc, false, false);
                }
                v8h out;
                #pragma unroll
                for (int r = 0; r < 8; ++r) {
                    float v = acc[r] + bias[r];
                    out[r] = (_Float16)(v > 0.f ? v : 0.f);
                }
                *(v8h*)&bufB[(nt * 16 + nSub) * ROWP + mBase] = out;
            }
        }
        __syncthreads();
        // ---- half 2: bufA = relu(bufA + W2 @ bufB + b2)  (in-place safe:
        //      only wave mt touches rows mBase.., residual read precedes write) ----
        {
            const _Float16* wmat = wpack + (size_t)(2 * d + 1) * 16384;
            if (d + 1 < DEPTH) {
                const _Float16* wnext = wpack + (size_t)(2 * d + 2) * 16384;
                #pragma unroll
                for (int kt = 0; kt < 4; ++kt)
                    __builtin_prefetch(wnext + ((mt * 4 + kt) * 512) + lane * 16, 0, 0);
            }
            v16h aF[4];
            #pragma unroll
            for (int kt = 0; kt < 4; ++kt)
                aF[kt] = *(const v16h*)(wmat + ((mt * 4 + kt) * 512) + lane * 16);
            float bias[8];
            #pragma unroll
            for (int r = 0; r < 8; ++r) bias[r] = b2s[d * CH + mBase + r];

            for (int nt = 0; nt < 8; ++nt) {
                v8f acc = {};
                #pragma unroll
                for (int kt = 0; kt < 4; ++kt) {
                    const _Float16* src = &bufB[(nt * 16 + nSub) * ROWP + kt * 32 + kHalf];
                    v8h lo = *(const v8h*)(src);
                    v8h hi = *(const v8h*)(src + 16);
                    v16h bF;
                    #pragma unroll
                    for (int i = 0; i < 8; ++i) { bF[i] = lo[i]; bF[8 + i] = hi[i]; }
                    acc = __builtin_amdgcn_wmma_f32_16x16x32_f16(
                        false, aF[kt], false, bF, (short)0, acc, false, false);
                }
                _Float16* hptr = &bufA[(nt * 16 + nSub) * ROWP + mBase];
                v8h hOld = *(const v8h*)hptr;
                v8h out;
                #pragma unroll
                for (int r = 0; r < 8; ++r) {
                    float v = acc[r] + bias[r] + (float)hOld[r];
                    out[r] = (_Float16)(v > 0.f ? v : 0.f);
                }
                *(v8h*)hptr = out;
            }
        }
        __syncthreads();
    }

    // ---------- logits, attention weight, 9x9 Gram accumulation ----------
    // Threads 0..127 each own one point (waves 0..3, all 32 lanes active).
    if (tid < TILE_N) {
        int p = tid;
        const v8h* row = (const v8h*)&bufA[p * ROWP];   // 16B-aligned (ROWP*2=272)
        float acc = bfin[0];
        for (int c8 = 0; c8 < CH / 8; ++c8) {
            v8h hv = row[c8];
            #pragma unroll
            for (int i = 0; i < 8; ++i)
                acc += (float)hv[i] * Wfin[c8 * 8 + i];
        }
        float t = tanhf(acc);
        float w = t > 0.f ? t : 0.f;
        float x1 = xb[0 * NPTS + p0 + p], x2 = xb[1 * NPTS + p0 + p];
        float y1 = xb[2 * NPTS + p0 + p], y2 = xb[3 * NPTS + p0 + p];
        float X[9] = { y1 * x1, y1 * x2, y1, y2 * x1, y2 * x2, y2, x1, x2, 1.f };

        // wave32 butterfly reduction per Gram entry, one global atomic per wave
        #pragma unroll
        for (int i = 0; i < 9; ++i) {
            float wi = w * X[i];
            #pragma unroll
            for (int j = i; j < 9; ++j) {            // symmetric: j >= i only
                float v = wi * X[j];
                #pragma unroll
                for (int m = 16; m >= 1; m >>= 1)
                    v += __shfl_xor(v, m, 32);
                if (lane == 0) {
                    atomicAdd(&xwx[b * 81 + i * 9 + j], v);
                    if (j != i) atomicAdd(&xwx[b * 81 + j * 9 + i], v);
                }
            }
        }
    }
}

// Per-batch 9x9 symmetric eigensolve (cyclic Jacobi), smallest eigenvector,
// normalized. One thread per batch.
__global__ void LGR_eigen(const float* __restrict__ xwx, float* __restrict__ out) {
    int b = threadIdx.x;
    if (b >= B_BATCH) return;
    float A[9][9], V[9][9];
    for (int i = 0; i < 9; ++i)
        for (int j = 0; j < 9; ++j) {
            A[i][j] = xwx[b * 81 + i * 9 + j];
            V[i][j] = (i == j) ? 1.f : 0.f;
        }
    for (int sweep = 0; sweep < 15; ++sweep) {
        for (int p = 0; p < 9; ++p) {
            for (int q = p + 1; q < 9; ++q) {
                float apq = A[p][q];
                if (fabsf(apq) < 1e-20f) continue;
                float theta = 0.5f * (A[q][q] - A[p][p]) / apq;
                float t = ((theta >= 0.f) ? 1.f : -1.f) /
                          (fabsf(theta) + sqrtf(1.f + theta * theta));
                float c = 1.f / sqrtf(1.f + t * t);
                float s = t * c;
                for (int k = 0; k < 9; ++k) {
                    float akp = A[k][p], akq = A[k][q];
                    A[k][p] = c * akp - s * akq;
                    A[k][q] = s * akp + c * akq;
                }
                for (int k = 0; k < 9; ++k) {
                    float apk = A[p][k], aqk = A[q][k];
                    A[p][k] = c * apk - s * aqk;
                    A[q][k] = s * apk + c * aqk;
                }
                for (int k = 0; k < 9; ++k) {
                    float vkp = V[k][p], vkq = V[k][q];
                    V[k][p] = c * vkp - s * vkq;
                    V[k][q] = s * vkp + c * vkq;
                }
            }
        }
    }
    int mi = 0; float mv = A[0][0];
    for (int i = 1; i < 9; ++i)
        if (A[i][i] < mv) { mv = A[i][i]; mi = i; }
    float nrm = 0.f;
    for (int i = 0; i < 9; ++i) nrm += V[i][mi] * V[i][mi];
    nrm = sqrtf(nrm);
    if (nrm < 1e-30f) nrm = 1.f;
    for (int i = 0; i < 9; ++i) out[b * 9 + i] = V[i][mi] / nrm;
}

extern "C" void kernel_launch(void* const* d_in, const int* in_sizes, int n_in,
                              void* d_out, int out_size, void* d_ws, size_t ws_size,
                              hipStream_t stream) {
    const float* x    = (const float*)d_in[0];
    const float* Wf   = (const float*)d_in[1];
    const float* bf   = (const float*)d_in[2];
    const float* W1s  = (const float*)d_in[3];
    const float* b1s  = (const float*)d_in[4];
    const float* W2s  = (const float*)d_in[5];
    const float* b2s  = (const float*)d_in[6];
    const float* Wfin = (const float*)d_in[7];
    const float* bfin = (const float*)d_in[8];

    _Float16* wpack = (_Float16*)d_ws;
    float* xwx = (float*)((char*)d_ws + XWX_OFF);

    hipMemsetAsync(xwx, 0, XWX_ELEMS * sizeof(float), stream);
    LGR_pack_weights<<<(WPACK_ELEMS + 255) / 256, 256, 0, stream>>>(W1s, W2s, wpack);
    LGR_mlp_xwx<<<B_BATCH * (NPTS / TILE_N), THREADS, 0, stream>>>(
        x, Wf, bf, b1s, b2s, Wfin, bfin, wpack, xwx);
    LGR_eigen<<<1, 64, 0, stream>>>(xwx, (float*)d_out);
}
